// Memory_multimode_41291815584281
// MI455X (gfx1250) — compile-verified, matrix operand order checked by reading, not run on hardware
//
#include <hip/hip_runtime.h>

// ---------------------------------------------------------------------------
// MI455X (gfx1250, wave32) fused 4-branch column-softmax attention.
//
// Math: Out[b] = (sum_c softmax_col(Q_c K_c)) @ V ; out = concat(Out, q_val)
//   Q in {q_key, p_q_key}  [1024 x 128]  (raw-reshape => already row-major)
//   K in {m_key, p_m_key}  [128 x 2048]
//   V = m_val flat          [2048 x 512]  (V[v*2048+j] => B-operand friendly)
// Shared-V fusion: one PV GEMM on the summed softmax instead of four.
//
// All matmuls via v_wmma_f32_16x16x32_bf16 (fp32 accum, fp32 softmax stats).
// K-tile staging in the attention hot loop uses the Tensor Data Mover
// (tensor_load_to_lds) with LDS pad insertion matching our 272B pitch.
//
// Workspace layout (bytes), total ~29.9 MB:
//   Qb    @ 0         : bf16 [2][8][1024][128]   (4 MB)
//   Kb    @ 4194304   : bf16 [2][8][2048][128]   (8 MB, K transposed)
//   Vb    @ 12582912  : bf16 [8][512][2048]      (16 MB)
//   stats @ 29360128  : f32x2 [4][8][2048]       (0.5 MB)  (colmax, 1/colsum)
// ---------------------------------------------------------------------------

typedef __attribute__((ext_vector_type(16))) __bf16 v16bf;
typedef __attribute__((ext_vector_type(8)))  float  v8f;
typedef __attribute__((ext_vector_type(4)))  unsigned int u32x4;
typedef __attribute__((ext_vector_type(8)))  unsigned int u32x8;
typedef __attribute__((ext_vector_type(4)))  int i32x4;
typedef __attribute__((ext_vector_type(8)))  int i32x8;

#define NB   8      // batches
#define HWN  1024   // H*W
#define KD   128    // keydim (contraction of GEMM1)
#define JD   2048   // 2*H*W  (softmax columns / contraction of GEMM2)
#define VD   512    // valdim

__device__ __forceinline__ unsigned short f2bf(float f) {
  unsigned int u = __float_as_uint(f);
  if ((u & 0x7fffffffu) > 0x7f800000u) return (unsigned short)((u >> 16) | 0x0040u); // NaN
  u += 0x7fffu + ((u >> 16) & 1u);                                                   // RNE
  return (unsigned short)(u >> 16);
}

// Build a WMMA 16-bit A/B operand from two 16-byte chunks (ISA 7.12.2 layout:
// per lane, bf16 elements [0..7] from p0 and [8..15] from p1 = base+32B).
__device__ __forceinline__ v16bf load_ab16(const void* p0, const void* p1) {
  u32x4 lo = *(const u32x4*)p0;
  u32x4 hi = *(const u32x4*)p1;
  u32x8 t;
  t[0] = lo[0]; t[1] = lo[1]; t[2] = lo[2]; t[3] = lo[3];
  t[4] = hi[0]; t[5] = hi[1]; t[6] = hi[2]; t[7] = hi[3];
  return __builtin_bit_cast(v16bf, t);
}

__device__ __forceinline__ v8f wmma_bf16(v16bf a, v16bf b, v8f c) {
  return __builtin_amdgcn_wmma_f32_16x16x32_bf16(false, a, false, b, (short)0, c,
                                                 false, false);
}

__device__ __forceinline__ v8f v8f_zero() {
  v8f z = {0.f, 0.f, 0.f, 0.f, 0.f, 0.f, 0.f, 0.f};
  return z;
}

// --------------------- Tensor Data Mover (TDM) helper ----------------------
// 2D bf16 tile load Global->LDS. D# per CDNA5 ISA:
//  group0: count=1 | lds_addr | global_addr(57b) | type=2
//  group1: data_size=2B, pad_enable, pad_interval/amount (encoded), dims/strides
// pad codes: interval 5 => 64 DWORDs (256B row), amount 3 => 4 DWORDs (16B).
__device__ __forceinline__ void tdm_load_2d_bf16(
    unsigned lds_off, const void* gptr,
    unsigned tensor_d0, unsigned tensor_d1,
    unsigned tile_d0, unsigned tile_d1,
    unsigned stride_d0,
    unsigned pad_interval_code, unsigned pad_amount_code, bool pad_en) {
  const unsigned long long ga = (unsigned long long)(size_t)gptr;
  u32x4 g0;
  g0[0] = 1u;                                            // count=1, user D#
  g0[1] = lds_off;                                       // lds_addr
  g0[2] = (unsigned)(ga & 0xffffffffull);                // global_addr[31:0]
  g0[3] = (unsigned)((ga >> 32) & 0x1ffffffull) | (2u << 30);  // [56:32]|type=2
  i32x8 g1;
  unsigned w0 = (1u << 16);                              // data_size = 2 bytes
  if (pad_en) w0 |= (1u << 20) | (pad_interval_code << 22) | (pad_amount_code << 25);
  g1[0] = (int)w0;
  g1[1] = (int)((tensor_d0 & 0xffffu) << 16);            // abar=0 | dim0[15:0]
  g1[2] = (int)(((tensor_d0 >> 16) & 0xffffu) | ((tensor_d1 & 0xffffu) << 16));
  g1[3] = (int)(((tensor_d1 >> 16) & 0xffffu) | ((tile_d0 & 0xffffu) << 16));
  g1[4] = (int)(tile_d1 & 0xffffu);                      // tile_dim1 | tile_dim2=0
  g1[5] = (int)stride_d0;                                // dim0_stride[31:0]
  g1[6] = 0;
  g1[7] = 0;
  const i32x4 z4 = {0, 0, 0, 0};
#if __clang_major__ >= 23
  const i32x8 z8 = {0, 0, 0, 0, 0, 0, 0, 0};
  __builtin_amdgcn_tensor_load_to_lds(g0, g1, z4, z4, z8, 0);
#else
  __builtin_amdgcn_tensor_load_to_lds(g0, g1, z4, z4, 0);
#endif
}

// ------------------------------ packing ------------------------------------

__global__ __launch_bounds__(256) void pack_q_kernel(
    const float* __restrict__ q_key, const float* __restrict__ p_q_key,
    unsigned short* __restrict__ Qb) {
  const int idx = blockIdx.x * 256 + threadIdx.x;      // < 2 * 1048576
  const int t   = idx >> 20;
  const int rem = idx & 0xFFFFF;
  const float v = (t == 0) ? q_key[rem] : p_q_key[rem];
  Qb[idx] = f2bf(v);
}

__global__ __launch_bounds__(256) void pack_v_kernel(
    const float* __restrict__ m_val, unsigned short* __restrict__ Vb) {
  const int idx = blockIdx.x * 256 + threadIdx.x;      // < 8388608
  Vb[idx] = f2bf(m_val[idx]);
}

// K: [128 x 2048] row-major -> Kb[j][kk] (transpose) via LDS tiles.
__global__ __launch_bounds__(256) void pack_k_kernel(
    const float* __restrict__ m_key, const float* __restrict__ p_m_key,
    unsigned short* __restrict__ Kb) {
  __shared__ float tile[32][33];
  const int jj0 = blockIdx.x * 32;
  const int kk0 = blockIdx.y * 32;
  const int tb  = blockIdx.z;
  const int t = tb >> 3, b = tb & 7;
  const float* src = ((t == 0) ? m_key : p_m_key) + (size_t)b * (KD * JD);
  const int tx = threadIdx.x, ty = threadIdx.y;
#pragma unroll
  for (int r = 0; r < 4; ++r)
    tile[ty + r * 8][tx] = src[(size_t)(kk0 + ty + r * 8) * JD + jj0 + tx];
  __syncthreads();
  unsigned short* dst = Kb + (size_t)(t * NB + b) * (JD * KD);
#pragma unroll
  for (int r = 0; r < 4; ++r)
    dst[(size_t)(jj0 + ty + r * 8) * KD + kk0 + tx] = f2bf(tile[tx][ty + r * 8]);
}

__global__ __launch_bounds__(256) void copy_qval_kernel(
    const float* __restrict__ q_val, float* __restrict__ out) {
  const int idx = blockIdx.x * 256 + threadIdx.x;      // < 4194304
  const int b = idx >> 19;
  const int r = idx & 0x7FFFF;
  out[(size_t)b * 1048576 + 524288 + r] = q_val[idx];
}

// ------------------------- pass 1: column stats ----------------------------
// grid = (jblocks=16, B=8, combo=4), block = 256 (8 waves).
// Per block: L = Q_c[1024x128] x K_c[:,jbase:jbase+128]; online (max, sumexp)
// per column, merged across 64-row chunks.

__global__ __launch_bounds__(256) void stats_kernel(
    const unsigned short* __restrict__ Qb,
    const unsigned short* __restrict__ Kb,
    float2* __restrict__ stats) {
  __shared__ unsigned char ldsK[128 * 272];   // K tile, pitch 272B kills bank dup
  __shared__ unsigned char ldsQ[64 * 272];    // Q chunk
  __shared__ float pM[128 * 8];
  __shared__ float pS[128 * 8];
  __shared__ float chM[128];
  __shared__ float runM[128];
  __shared__ float runS[128];

  const int cmb   = blockIdx.z;
  const int b     = blockIdx.y;
  const int jbase = blockIdx.x * 128;
  const int qsel = cmb >> 1, ksel = cmb & 1;
  const int tid = threadIdx.x;
  const int lane = tid & 31;
  const int w = tid >> 5;
  const int mb = w & 3;
  const int nbBase = (w >> 2) * 4;
  const int lg = lane >> 4;   // lane group selects K-halves (ISA A/B layout)
  const int lr = lane & 15;

  {  // stage K tile [128 rows x 128 kk] bf16 (resident for whole block)
    const unsigned short* kp =
        Kb + (size_t)((ksel * NB + b) * JD + jbase) * KD;
    const int r = tid >> 1, half = tid & 1;
    const u32x4* src = (const u32x4*)(kp + r * KD + half * 64);
    u32x4* dst = (u32x4*)(ldsK + r * 272 + half * 128);
#pragma unroll
    for (int x = 0; x < 8; ++x) dst[x] = src[x];
  }
  if (tid < 128) { runM[tid] = -3.0e38f; runS[tid] = 0.0f; }

  for (int ic = 0; ic < 16; ++ic) {
    __syncthreads();  // prev chunk done with ldsQ/pM/pS/chM (+K visible 1st it)
    {  // stage Q chunk [64 rows x 128 kk]
      const unsigned short* qp =
          Qb + (size_t)((qsel * NB + b) * HWN + ic * 64) * KD;
      const int r = tid >> 2, qq = tid & 3;
      const u32x4* src = (const u32x4*)(qp + r * KD + qq * 32);
      u32x4* dst = (u32x4*)(ldsQ + r * 272 + qq * 64);
#pragma unroll
      for (int x = 0; x < 4; ++x) dst[x] = src[x];
    }
    __syncthreads();

    v8f L[4];
#pragma unroll
    for (int n = 0; n < 4; ++n) L[n] = v8f_zero();

#pragma unroll
    for (int kb = 0; kb < 4; ++kb) {
      const int aoff = (mb * 16 + lr) * 272 + kb * 64 + lg * 16;
      const v16bf a = load_ab16(ldsQ + aoff, ldsQ + aoff + 32);
#pragma unroll
      for (int n = 0; n < 4; ++n) {
        const int boff = ((nbBase + n) * 16 + lr) * 272 + kb * 64 + lg * 16;
        const v16bf bm = load_ab16(ldsK + boff, ldsK + boff + 32);
        L[n] = wmma_bf16(a, bm, L[n]);
      }
    }

    const int slot = mb * 2 + lg;
#pragma unroll
    for (int n = 0; n < 4; ++n) {  // chunk column maxes (8 slots per column)
      float m = L[n][0];
#pragma unroll
      for (int r = 1; r < 8; ++r) m = fmaxf(m, L[n][r]);
      pM[((nbBase + n) * 16 + lr) * 8 + slot] = m;
    }
    __syncthreads();
    if (tid < 128) {
      float m = pM[tid * 8];
#pragma unroll
      for (int s = 1; s < 8; ++s) m = fmaxf(m, pM[tid * 8 + s]);
      chM[tid] = m;
    }
    __syncthreads();
#pragma unroll
    for (int n = 0; n < 4; ++n) {  // chunk column sumexp
      const int col = (nbBase + n) * 16 + lr;
      const float m = chM[col];
      float s = 0.f;
#pragma unroll
      for (int r = 0; r < 8; ++r) s += __expf(L[n][r] - m);
      pS[col * 8 + slot] = s;
    }
    __syncthreads();
    if (tid < 128) {  // online merge into running (m, s)
      float s = 0.f;
#pragma unroll
      for (int k = 0; k < 8; ++k) s += pS[tid * 8 + k];
      const float mc = chM[tid];
      const float mo = runM[tid];
      const float mn = fmaxf(mo, mc);
      runS[tid] = runS[tid] * __expf(mo - mn) + s * __expf(mc - mn);
      runM[tid] = mn;
    }
  }
  __syncthreads();
  if (tid < 128) {
    const float z = runS[tid];
    stats[(size_t)(cmb * NB + b) * JD + jbase + tid] =
        make_float2(runM[tid], z > 0.f ? 1.f / z : 0.f);
  }
}

// ----------------- pass 2: recompute logits, S = sum_c softmax, @ V --------
// grid = (iblocks=16, vblocks=2, B=8), block = 256 (8 waves).
// Wave w: mb = w&3 (16-row stripe), grp = w>>2 (column half).
// Per 64-j step: wave 0 DMAs both K tiles into LDS via TDM (padded to 272B
// pitch), s_wait_tensorcnt, barrier; then 4 combo logit tiles (Q in regs) ->
// S bf16 via LDS -> 2x8 WMMA against V straight from global.

__global__ __launch_bounds__(256) void attn_kernel(
    const unsigned short* __restrict__ Qb,
    const unsigned short* __restrict__ Kb,
    const unsigned short* __restrict__ Vb,
    const float2* __restrict__ stats,
    float* __restrict__ out) {
  __shared__ unsigned char ldsK[2 * 64 * 272];  // 34816 B (Mk + Pk tiles)
  __shared__ unsigned char ldsS[64 * 144];      //  9216 B (S tile, bf16)

  const int b     = blockIdx.z;
  const int ibase = blockIdx.x * 64;
  const int vbase = blockIdx.y * 256;
  const int tid = threadIdx.x;
  const int lane = tid & 31;
  const int w = tid >> 5;
  const int mb = w & 3;
  const int grp = w >> 2;
  const int lg = lane >> 4;
  const int lr = lane & 15;
  const unsigned ldsK_off = (unsigned)(size_t)(void*)ldsK;  // LDS byte offset

  // Q A-operands resident in VGPRs: 2 tensors x 4 k-steps x 8 VGPRs.
  v16bf Qa[2][4];
#pragma unroll
  for (int qs = 0; qs < 2; ++qs) {
    const unsigned short* qp =
        Qb + (size_t)((qs * NB + b) * HWN + ibase + mb * 16 + lr) * KD;
#pragma unroll
    for (int kb = 0; kb < 4; ++kb) {
      const int off = kb * 32 + lg * 8;
      Qa[qs][kb] = load_ab16(qp + off, qp + off + 16);
    }
  }

  v8f C[8];
#pragma unroll
  for (int n = 0; n < 8; ++n) C[n] = v8f_zero();

  const unsigned short* VbB = Vb + (size_t)b * (VD * JD);

  for (int jt = 0; jt < 32; ++jt) {
    const int jbase = jt * 64;
    if (w == 0) {  // wave-uniform branch: TDM issues once per instruction
#pragma unroll
      for (int ts = 0; ts < 2; ++ts) {
        const unsigned short* kp =
            Kb + (size_t)((ts * NB + b) * JD + jbase) * KD;
        // tile: 128 elems (kk) x 64 rows (j); row stride 128 elems; LDS rows
        // padded 256B+16B (interval code 5 = 64 DW, amount code 3 = 4 DW).
        tdm_load_2d_bf16(ldsK_off + ts * 17408, kp,
                         /*tensor_d0=*/KD, /*tensor_d1=*/JD,
                         /*tile_d0=*/KD, /*tile_d1=*/64,
                         /*stride_d0=*/KD,
                         /*pad_interval=*/5u, /*pad_amount=*/3u, true);
      }
      __builtin_amdgcn_s_wait_tensorcnt(0);
    }
    __syncthreads();

#pragma unroll
    for (int nbi = 0; nbi < 2; ++nbi) {
      const int nb = grp * 2 + nbi;
      const int col = jbase + nb * 16 + lr;
      v8f S = v8f_zero();
#pragma unroll
      for (int cmb = 0; cmb < 4; ++cmb) {
        const int qs = cmb >> 1, ks = cmb & 1;
        v8f L = v8f_zero();
#pragma unroll
        for (int kb = 0; kb < 4; ++kb) {
          const int boff =
              ks * 17408 + (nb * 16 + lr) * 272 + kb * 64 + lg * 16;
          const v16bf bm = load_ab16(ldsK + boff, ldsK + boff + 32);
          L = wmma_bf16(Qa[qs][kb], bm, L);
        }
        const float2 st = stats[(size_t)(cmb * NB + b) * JD + col];
#pragma unroll
        for (int r = 0; r < 8; ++r) S[r] += __expf(L[r] - st.x) * st.y;
      }
      // C-layout -> row-major bf16 S tile in LDS (A-layout source for GEMM2)
      const int srow  = mb * 16 + lg * 8;
      const int scolb = (nb * 16 + lr) * 2;
#pragma unroll
      for (int r = 0; r < 8; ++r)
        *(unsigned short*)(ldsS + (srow + r) * 144 + scolb) = f2bf(S[r]);
    }
    __syncthreads();

#pragma unroll
    for (int kb2 = 0; kb2 < 2; ++kb2) {
      const int aoff = (mb * 16 + lr) * 144 + kb2 * 64 + lg * 16;
      const v16bf a = load_ab16(ldsS + aoff, ldsS + aoff + 32);
#pragma unroll
      for (int nj = 0; nj < 8; ++nj) {
        const int n = vbase + (grp * 8 + nj) * 16 + lr;
        const unsigned short* vp =
            VbB + (size_t)n * JD + jbase + kb2 * 32 + lg * 8;
        const v16bf bm = load_ab16(vp, vp + 16);
        if (kb2 == 0 && jt + 1 < 32)
          __builtin_prefetch(vp + 64, 0, 1);  // next j-tile of V
        C[nj] = wmma_bf16(a, bm, C[nj]);
      }
    }
    __syncthreads();  // S tile / K tiles reused next iteration
  }

  const size_t ob = (size_t)b * 1048576;  // batch slab in d_out
  const int rowb = ibase + mb * 16 + lg * 8;
#pragma unroll
  for (int nj = 0; nj < 8; ++nj) {
    const int v = vbase + (grp * 8 + nj) * 16 + lr;
#pragma unroll
    for (int r = 0; r < 8; ++r)
      out[ob + (size_t)(rowb + r) * VD + v] = C[nj][r];
  }
}

// ------------------------------- launch ------------------------------------

extern "C" void kernel_launch(void* const* d_in, const int* in_sizes, int n_in,
                              void* d_out, int out_size, void* d_ws,
                              size_t ws_size, hipStream_t stream) {
  const float* m_key   = (const float*)d_in[0];
  const float* m_val   = (const float*)d_in[1];
  const float* q_key   = (const float*)d_in[2];
  const float* q_val   = (const float*)d_in[3];
  const float* p_m_key = (const float*)d_in[4];
  const float* p_q_key = (const float*)d_in[5];
  float* out = (float*)d_out;

  char* ws = (char*)d_ws;                       // needs ~29.9 MB
  unsigned short* Qb = (unsigned short*)(ws + 0);
  unsigned short* Kb = (unsigned short*)(ws + 4194304);
  unsigned short* Vb = (unsigned short*)(ws + 12582912);
  float2* stats      = (float2*)(ws + 29360128);

  pack_q_kernel<<<8192, 256, 0, stream>>>(q_key, p_q_key, Qb);
  pack_v_kernel<<<32768, 256, 0, stream>>>(m_val, Vb);
  pack_k_kernel<<<dim3(64, 4, 16), dim3(32, 8), 0, stream>>>(m_key, p_m_key, Kb);
  copy_qval_kernel<<<16384, 256, 0, stream>>>(q_val, out);

  stats_kernel<<<dim3(16, NB, 4), 256, 0, stream>>>(Qb, Kb, stats);
  attn_kernel<<<dim3(16, 2, NB), 256, 0, stream>>>(Qb, Kb, Vb, stats, out);
}
